// block_55035710931798
// MI455X (gfx1250) — matrix operand order repaired and checked by reading, not compiled, not run on hardware
//
#include <hip/hip_runtime.h>
#include <hip/hip_bf16.h>
#include <math.h>

#define BB   2
#define TDIM 2048
#define CDIM 1024
#define HH   16
#define DD   64
#define BT   (BB * TDIM)   // 4096 tokens

typedef __attribute__((ext_vector_type(16))) __bf16 v16bf;
typedef __attribute__((ext_vector_type(8)))  float  v8f;

// ---------------------------------------------------------------------------
// WMMA helpers (CDNA5: D = A(16x32 bf16) * B(32x16 bf16) + C(16x16 f32))
// ---------------------------------------------------------------------------
__device__ __forceinline__ v8f wmma_bf16(v16bf a, v16bf b, v8f c) {
  return __builtin_amdgcn_wmma_f32_16x16x32_bf16(
      /*neg_a=*/false, a, /*neg_b=*/false, b,
      /*c_mod=*/(short)0, c, /*reuse_a=*/false, /*reuse_b=*/false);
}

// A fragment from a row-major [M][K] tile (base already at (m0,k0)).
// Layout: lanes 0-15 -> M=lane, K = 0..15
//         lanes 16-31 -> M=lane-16, K = 16..31
// One contiguous 32B run per lane (matches swizzled B: elem = k&15).
__device__ __forceinline__ v16bf load_a_frag(const __bf16* base, int ld, int lane) {
  const int m  = lane & 15;
  const int kk = (lane & 16) ? 16 : 0;
  const __bf16* r = base + (size_t)m * ld + kk;
  v16bf a;
#pragma unroll
  for (int j = 0; j < 8; ++j) a[j] = r[j];
#pragma unroll
  for (int j = 0; j < 8; ++j) a[8 + j] = r[8 + j];
  return a;
}

// B fragment from a pre-swizzled tile: layout [n_tile][32 lanes][16 elems],
// where for element (k, n) of a 32xN tile:
//   n_tile = n>>4, lane = (n&15) + 16*(k>=16), elem = k&15.
// Per-lane read is one contiguous 32B -> 2x ds_load_b128, no packing VALU.
__device__ __forceinline__ v16bf load_b_frag_sw(const __bf16* base, int lane) {
  return *(const v16bf*)(base + lane * 16);
}

// Swizzled offset helper for writers.
__device__ __forceinline__ int bswz(int k, int n) {
  return (n >> 4) * 512 + (((n & 15) + ((k >> 4) << 4)) << 4) + (k & 15);
}

// ---------------------------------------------------------------------------
// LayerNorm (per token, C=1024) -> bf16
// ---------------------------------------------------------------------------
__global__ __launch_bounds__(256) void ln_bf16_kernel(
    const float* __restrict__ x, const float* __restrict__ g,
    const float* __restrict__ bta, __bf16* __restrict__ out) {
  __shared__ float red[256];
  const int row = blockIdx.x;
  const int tid = threadIdx.x;
  const float* xr = x + (size_t)row * CDIM;

  float v0 = xr[tid], v1 = xr[tid + 256], v2 = xr[tid + 512], v3 = xr[tid + 768];
  red[tid] = v0 + v1 + v2 + v3;
  __syncthreads();
#pragma unroll
  for (int off = 128; off > 0; off >>= 1) {
    if (tid < off) red[tid] += red[tid + off];
    __syncthreads();
  }
  const float mu = red[0] * (1.0f / CDIM);
  __syncthreads();

  float d0 = v0 - mu, d1 = v1 - mu, d2 = v2 - mu, d3 = v3 - mu;
  red[tid] = d0 * d0 + d1 * d1 + d2 * d2 + d3 * d3;
  __syncthreads();
#pragma unroll
  for (int off = 128; off > 0; off >>= 1) {
    if (tid < off) red[tid] += red[tid + off];
    __syncthreads();
  }
  const float rstd = rsqrtf(red[0] * (1.0f / CDIM) + 1e-5f);

  __bf16* o = out + (size_t)row * CDIM;
  o[tid]       = (__bf16)(d0 * rstd * g[tid]       + bta[tid]);
  o[tid + 256] = (__bf16)(d1 * rstd * g[tid + 256] + bta[tid + 256]);
  o[tid + 512] = (__bf16)(d2 * rstd * g[tid + 512] + bta[tid + 512]);
  o[tid + 768] = (__bf16)(d3 * rstd * g[tid + 768] + bta[tid + 768]);
}

// ---------------------------------------------------------------------------
// GEMM: out[M,N] = epilogue( A_bf16[M,K] @ W_f32[K,N] + bias )
// Block tile 128x128, BK=32, 8 waves each computing a 32x64 sub-tile
// (2x4 WMMA tiles => 8 v_wmma per wave per K-step).
// ---------------------------------------------------------------------------
__global__ __launch_bounds__(256) void gemm_wmma_kernel(
    const __bf16* __restrict__ A, const float* __restrict__ W,
    const float* __restrict__ bias, const float* __restrict__ resid,
    float* __restrict__ outF, __bf16* __restrict__ outB,
    int M, int N, int K, int act) {
  __shared__ __align__(32) __bf16 As[128 * 32];   // row-major [m][k]
  __shared__ __align__(32) __bf16 Bs[8 * 512];    // swizzled: 8 n-tiles of 32x16

  const int tid  = threadIdx.x;
  const int wave = tid >> 5;
  const int lane = tid & 31;
  const int m0 = blockIdx.y * 128;
  const int n0 = blockIdx.x * 128;
  const int wm = (wave >> 1) * 32;  // wave sub-tile row  (0,32,64,96)
  const int wn = (wave & 1) * 64;   // wave sub-tile col  (0,64)
  const int wnt = wn >> 4;          // first n-tile index for this wave

  v8f acc[2][4] = {};

  const int ar  = tid >> 1;         // 0..127 : A tile row
  const int ak  = (tid & 1) * 16;   // A tile k-half
  const int wr  = tid >> 3;         // 0..31  : W tile row (k)
  const int wc0 = (tid & 7) * 16;   // W tile col group (16 wide)
  const int whi = (wr >> 4) << 4;   // swizzle lane-half from k
  const int wel = wr & 15;          // swizzle element from k

  for (int kb = 0; kb < K; kb += 32) {
    // A tile: bf16, contiguous copy
    const __bf16* ag = A + (size_t)(m0 + ar) * K + kb + ak;
#pragma unroll
    for (int j = 0; j < 16; ++j) As[ar * 32 + ak + j] = ag[j];

    // W tile: f32 -> bf16, scatter into fragment-swizzled layout
    const float* wg = W + (size_t)(kb + wr) * N + n0 + wc0;
#pragma unroll
    for (int j = 0; j < 16; ++j) {
      const int n = wc0 + j;
      Bs[(n >> 4) * 512 + (((n & 15) + whi) << 4) + wel] = (__bf16)wg[j];
    }

    if (kb + 32 < K)  // hint next weight tile (global_prefetch_b8)
      __builtin_prefetch(W + (size_t)(kb + 32 + wr) * N + n0 + wc0, 0, 1);

    __syncthreads();

    v16bf a0 = load_a_frag(As + (wm + 0) * 32, 32, lane);
    v16bf a1 = load_a_frag(As + (wm + 16) * 32, 32, lane);
#pragma unroll
    for (int jn = 0; jn < 4; ++jn) {
      v16bf b = load_b_frag_sw(Bs + (wnt + jn) * 512, lane);
      acc[0][jn] = wmma_bf16(a0, b, acc[0][jn]);
      acc[1][jn] = wmma_bf16(a1, b, acc[1][jn]);
    }
    __syncthreads();
  }

  // Epilogue. C/D layout: lane -> col = lane&15, rows = (lane>=16 ? 8..15 : 0..7)
  const int cl = lane & 15;
  const int r0 = (lane >> 4) * 8;
#pragma unroll
  for (int im = 0; im < 2; ++im) {
#pragma unroll
    for (int jn = 0; jn < 4; ++jn) {
      const int col = n0 + wn + jn * 16 + cl;
      const float bval = bias[col];
#pragma unroll
      for (int j = 0; j < 8; ++j) {
        const int row = m0 + wm + im * 16 + r0 + j;
        float vv = acc[im][jn][j] + bval;
        if (act == 1) vv = 0.5f * vv * (1.0f + erff(vv * 0.70710678118654752f));
        if (resid) vv += resid[(size_t)row * N + col];
        const size_t idx = (size_t)row * N + col;
        if (outF) outF[idx] = vv;
        if (outB) outB[idx] = (__bf16)vv;
      }
    }
  }
}

// ---------------------------------------------------------------------------
// Flash attention: q,k,v bf16 [B,T,H,D] -> y bf16 [B,T,C]
// Block = 128 query rows of one (b,h); 8 waves x 16 rows. 64-key tiles.
// Kt/Vt are stored in fragment-swizzled layout: [ks][nt][32 lanes][16 elems].
// ---------------------------------------------------------------------------
__global__ __launch_bounds__(256) void attn_kernel(
    const __bf16* __restrict__ q, const __bf16* __restrict__ k,
    const __bf16* __restrict__ v, __bf16* __restrict__ y) {
  __shared__ __align__(32) __bf16 Kt[2 * 2048];     // B for S = Q K^T (K-dim = d)
  __shared__ __align__(32) __bf16 Vt[2 * 2048];     // B for O = P V   (K-dim = key)
  __shared__ __align__(32) __bf16 Pls[8][16 * 64];  // per-wave P staging

  const int tid  = threadIdx.x;
  const int wave = tid >> 5;
  const int lane = tid & 31;
  const int b = blockIdx.z, h = blockIdx.y;
  const int q0 = blockIdx.x * 128 + wave * 16;

  const int ldq = HH * DD;
  const __bf16* qbase = q + ((size_t)(b * TDIM + q0) * HH + h) * DD;
  v16bf aq0 = load_a_frag(qbase, ldq, lane);       // K dims 0..31
  v16bf aq1 = load_a_frag(qbase + 32, ldq, lane);  // K dims 32..63

  float mrow[8], lrow[8];
  v8f o[4] = {};
#pragma unroll
  for (int j = 0; j < 8; ++j) { mrow[j] = -1e30f; lrow[j] = 0.0f; }

  const int ls  = tid >> 2;        // 0..63 key index for loads
  const int ld0 = (tid & 3) * 16;  // d-group for loads
  const int cl  = lane & 15;
  const int r0  = (lane >> 4) * 8;
  __bf16* Pw = &Pls[wave][0];

  // Kt writer: k-dim = d (spans ld0..ld0+15, one half), n = key ls
  //   -> swizzled destination is CONTIGUOUS: a straight 32B vector copy.
  const int kks  = ld0 >> 5;                       // d sub-tile (K step)
  const int khi  = ((ld0 & 31) >> 4) << 4;         // lane half from d
  __bf16* ktdst = &Kt[kks * 2048 + (ls >> 4) * 512 + (((ls & 15) + khi) << 4)];
  // Vt writer: k-dim = key ls, n = d (ld0..ld0+15) -> 16-store scatter
  const int vks = ls >> 5;
  const int vhi = ((ls & 31) >> 4) << 4;
  const int vel = ls & 15;
  __bf16* vtbase = &Vt[vks * 2048 + (ld0 >> 4) * 512];

  for (int kt = 0; kt < TDIM / 64; ++kt) {
    const int s0 = kt * 64;
    const __bf16* kg = k + ((size_t)(b * TDIM + s0 + ls) * HH + h) * DD + ld0;
    const __bf16* vg = v + ((size_t)(b * TDIM + s0 + ls) * HH + h) * DD + ld0;
    *(v16bf*)ktdst = *(const v16bf*)kg;            // transposed via swizzle
#pragma unroll
    for (int j = 0; j < 16; ++j)
      vtbase[((j + vhi) << 4) + vel] = vg[j];
    __syncthreads();

    // S = (Q K^T) * 1/sqrt(D): 16x64 per wave = 4 n-tiles
    v8f st[4];
#pragma unroll
    for (int nt = 0; nt < 4; ++nt) {
      v8f c = {};
      c = wmma_bf16(aq0, load_b_frag_sw(Kt + nt * 512, lane), c);
      c = wmma_bf16(aq1, load_b_frag_sw(Kt + 2048 + nt * 512, lane), c);
      st[nt] = c * 0.125f;
    }

    // row max across 4 tiles (in-lane) then across the 16-lane half
    float mt[8];
#pragma unroll
    for (int j = 0; j < 8; ++j)
      mt[j] = fmaxf(fmaxf(st[0][j], st[1][j]), fmaxf(st[2][j], st[3][j]));
#pragma unroll
    for (int mask = 1; mask < 16; mask <<= 1)
#pragma unroll
      for (int j = 0; j < 8; ++j) mt[j] = fmaxf(mt[j], __shfl_xor(mt[j], mask, 32));

    float sf[8], mnew[8], rs[8];
#pragma unroll
    for (int j = 0; j < 8; ++j) {
      mnew[j] = fmaxf(mrow[j], mt[j]);
      sf[j]   = __expf(mrow[j] - mnew[j]);
      rs[j]   = 0.0f;
    }
#pragma unroll
    for (int nt = 0; nt < 4; ++nt)
#pragma unroll
      for (int j = 0; j < 8; ++j) {
        const float p = __expf(st[nt][j] - mnew[j]);
        st[nt][j] = p;
        rs[j] += p;
      }
#pragma unroll
    for (int mask = 1; mask < 16; mask <<= 1)
#pragma unroll
      for (int j = 0; j < 8; ++j) rs[j] += __shfl_xor(rs[j], mask, 32);
#pragma unroll
    for (int j = 0; j < 8; ++j) {
      lrow[j] = lrow[j] * sf[j] + rs[j];
      mrow[j] = mnew[j];
    }
#pragma unroll
    for (int nt = 0; nt < 4; ++nt)
#pragma unroll
      for (int j = 0; j < 8; ++j) o[nt][j] *= sf[j];

    // Stage P (D-layout) into per-wave LDS strip, reload in A-layout.
#pragma unroll
    for (int nt = 0; nt < 4; ++nt)
#pragma unroll
      for (int j = 0; j < 8; ++j)
        Pw[(r0 + j) * 64 + nt * 16 + cl] = (__bf16)st[nt][j];

    v16bf pa0 = load_a_frag(Pw, 64, lane);
    v16bf pa1 = load_a_frag(Pw + 32, 64, lane);
#pragma unroll
    for (int nt = 0; nt < 4; ++nt) {
      o[nt] = wmma_bf16(pa0, load_b_frag_sw(Vt + nt * 512, lane), o[nt]);
      o[nt] = wmma_bf16(pa1, load_b_frag_sw(Vt + 2048 + nt * 512, lane), o[nt]);
    }
    __syncthreads();
  }

  // y[b, row, h*D + d] = O / l
#pragma unroll
  for (int nt = 0; nt < 4; ++nt)
#pragma unroll
    for (int j = 0; j < 8; ++j) {
      const int row = q0 + r0 + j;
      const float val = o[nt][j] / lrow[j];
      y[((size_t)(b * TDIM + row) * HH + h) * DD + nt * 16 + cl] = (__bf16)val;
    }
}

// ---------------------------------------------------------------------------
// Launch: LN1 -> QKV GEMMs -> flash attn -> Wo(+resid) -> LN2 -> W1(GELU) -> W2(+resid)
// ---------------------------------------------------------------------------
extern "C" void kernel_launch(void* const* d_in, const int* in_sizes, int n_in,
                              void* d_out, int out_size, void* d_ws, size_t ws_size,
                              hipStream_t stream) {
  (void)in_sizes; (void)n_in; (void)out_size; (void)ws_size;
  const float* x   = (const float*)d_in[0];
  const float* Wq  = (const float*)d_in[1];
  const float* bq  = (const float*)d_in[2];
  const float* Wk  = (const float*)d_in[3];
  const float* bk  = (const float*)d_in[4];
  const float* Wv  = (const float*)d_in[5];
  const float* bv  = (const float*)d_in[6];
  const float* Wo  = (const float*)d_in[7];
  const float* bo  = (const float*)d_in[8];
  const float* g1  = (const float*)d_in[9];
  const float* be1 = (const float*)d_in[10];
  const float* g2  = (const float*)d_in[11];
  const float* be2 = (const float*)d_in[12];
  const float* W1  = (const float*)d_in[13];
  const float* b1  = (const float*)d_in[14];
  const float* W2  = (const float*)d_in[15];
  const float* b2  = (const float*)d_in[16];

  char* ws = (char*)d_ws;
  const size_t MB = 1024ull * 1024ull;
  __bf16* hB = (__bf16*)(ws);             //  8 MB: LN1/LN2 output (reused)
  __bf16* qB = (__bf16*)(ws +  8 * MB);   //  8 MB
  __bf16* kB = (__bf16*)(ws + 16 * MB);   //  8 MB
  __bf16* vB = (__bf16*)(ws + 24 * MB);   //  8 MB
  __bf16* yB = (__bf16*)(ws + 32 * MB);   //  8 MB: attention output
  float*  x1 = (float*)(ws + 40 * MB);    // 16 MB: post-attention residual
  __bf16* mB = (__bf16*)(ws + 56 * MB);   // 32 MB: GELU(h2 W1 + b1)

  dim3 blk(256);
  dim3 gC(CDIM / 128, BT / 128);         // N=1024 GEMMs
  dim3 gM(4 * CDIM / 128, BT / 128);     // N=4096 GEMM
  dim3 gA(TDIM / 128, HH, BB);

  ln_bf16_kernel<<<BT, blk, 0, stream>>>(x, g1, be1, hB);

  gemm_wmma_kernel<<<gC, blk, 0, stream>>>(hB, Wq, bq, nullptr, nullptr, qB,
                                           BT, CDIM, CDIM, 0);
  gemm_wmma_kernel<<<gC, blk, 0, stream>>>(hB, Wk, bk, nullptr, nullptr, kB,
                                           BT, CDIM, CDIM, 0);
  gemm_wmma_kernel<<<gC, blk, 0, stream>>>(hB, Wv, bv, nullptr, nullptr, vB,
                                           BT, CDIM, CDIM, 0);

  attn_kernel<<<gA, blk, 0, stream>>>(qB, kB, vB, yB);

  gemm_wmma_kernel<<<gC, blk, 0, stream>>>(yB, Wo, bo, x, x1, nullptr,
                                           BT, CDIM, CDIM, 0);

  ln_bf16_kernel<<<BT, blk, 0, stream>>>(x1, g2, be2, hB);

  gemm_wmma_kernel<<<gM, blk, 0, stream>>>(hB, W1, b1, nullptr, nullptr, mB,
                                           BT, 4 * CDIM, CDIM, 1);
  gemm_wmma_kernel<<<gC, blk, 0, stream>>>(mB, W2, b2, x1, (float*)d_out, nullptr,
                                           BT, CDIM, 4 * CDIM, 0);
}
